// LSTM_Classifier_20607253086531
// MI455X (gfx1250) — matrix operand order repaired and checked by reading, not compiled, not run on hardware
//
#include <hip/hip_runtime.h>
#include <hip/hip_bf16.h>

// ---------------------------------------------------------------------------
// LSTM classifier on MI455X (gfx1250), bf16 WMMA path + TDM weight staging.
//   B=256, T=128, IN=512, EMB=HID=1024, 2 layers, 10 classes.
// Weights are converted to bf16 and pre-swizzled into WMMA B-fragment order;
// each (t,layer) step is ONE kernel fusing both GEMMs + the full LSTM cell.
// Weight panels stream into LDS via the Tensor Data Mover (tensor_load_to_lds,
// s_wait_tensorcnt), so the inner loop is pure ds_load + v_wmma.
// ---------------------------------------------------------------------------

#define BSZ    256
#define TSTEPS 128
#define INDIM  512
#define EMB    1024
#define HID    1024

typedef __bf16 bf16;
typedef __attribute__((ext_vector_type(16))) __bf16 v16bf;
typedef __attribute__((ext_vector_type(8)))  __bf16 v8bf;
typedef __attribute__((ext_vector_type(8)))  float  v8f;
typedef __attribute__((ext_vector_type(4)))  unsigned int v4u;
typedef __attribute__((ext_vector_type(4)))  int v4i;
typedef __attribute__((ext_vector_type(8)))  int v8i;

union V16 { v16bf v; v8bf h[2]; };

__device__ inline float sigf(float x) { return 1.0f / (1.0f + __expf(-x)); }
__device__ inline float tanhf_fast(float x) {
  float e = __expf(2.0f * x);
  return (e - 1.0f) / (e + 1.0f);
}

__device__ inline v8f wmma_bf16(v16bf a, v16bf b, v8f c) {
  // D = A(16x32) x B(32x16) + C, f32 accumulation
  return __builtin_amdgcn_wmma_f32_16x16x32_bf16(
      false, a, false, b, (short)0, c, false, false);
}

// A fragment (16x32 bf16), row-major source, row = lane%16:
//   half = lane/16: lane covers K in [16*half,16*half+16)
__device__ inline v16bf load_a(const bf16* __restrict__ rowptr, int k, int half) {
  V16 u;
  u.h[0] = *(const v8bf*)(rowptr + k + 16 * half);
  u.h[1] = *(const v8bf*)(rowptr + k + 16 * half + 8);
  return u.v;
}

// ---------------------------------------------------------------------------
// TDM: 1-D strip load global->LDS.  nElem8 = number of 8-byte elements.
// D# packing per CDNA5 ISA (groups 0 and 1; groups 2/3
// zero => <=2D tensor).  All operands are wave-uniform -> SGPR descriptors.
// ---------------------------------------------------------------------------
__device__ inline void tdm_load_strip(const void* gsrc, unsigned ldsByteOff,
                                      unsigned nElem8) {
  unsigned long long ga = (unsigned long long)gsrc;
  v4u g0;
  g0.x = 1u;                                   // count=1 (valid user desc)
  g0.y = ldsByteOff;                           // lds_addr (bytes)
  g0.z = (unsigned)ga;                         // global_addr[31:0]
  g0.w = (unsigned)((ga >> 32) & 0x01FFFFFFu)  // global_addr[56:32]
         | 0x80000000u;                        // type=2 ("image") at [127:126]
  v8i g1;
  g1[0] = 0x00030000;                          // data_size=3 (8B), mask=0
  g1[1] = (int)(nElem8 << 16);                 // tensor_dim0[15:0]
  g1[2] = (int)(1u << 16);                     // tensor_dim0 hi=0 | tensor_dim1=1
  g1[3] = (int)(nElem8 << 16);                 // tensor_dim1 hi=0 | tile_dim0
  g1[4] = 0;                                   // tile_dim1=0 (1-D), tile_dim2=0
  g1[5] = (int)nElem8;                         // tensor_dim0_stride[31:0]
  g1[6] = 0;
  g1[7] = 0;
  v4i z4 = {0, 0, 0, 0};
#if __clang_major__ >= 23
  v8i z8 = {0, 0, 0, 0, 0, 0, 0, 0};
  __builtin_amdgcn_tensor_load_to_lds(g0, g1, z4, z4, z8, 0);
#else
  __builtin_amdgcn_tensor_load_to_lds(g0, g1, z4, z4, 0);
#endif
}

// ---------------------------------------------------------------------------
// Prep: f32 -> bf16 elementwise
// ---------------------------------------------------------------------------
__global__ void cvt_bf16_kernel(const float* __restrict__ in,
                                bf16* __restrict__ out, int n) {
  int tid = blockIdx.x * blockDim.x + threadIdx.x;
  if (tid < n) out[tid] = (bf16)in[tid];
}

// ---------------------------------------------------------------------------
// Prep: swizzle W [N,K] row-major f32 into WMMA B-fragment order (bf16):
//   out[((nblk*(K/32)+kblk)*32 + lane)*16 + e] = W[nblk*16 + lane%16,
//                                                  kblk*32 + (lane/16)*16 + e]
// ---------------------------------------------------------------------------
__global__ void swizzle_weight_kernel(const float* __restrict__ W,
                                      bf16* __restrict__ out, int N, int K) {
  int tid = blockIdx.x * blockDim.x + threadIdx.x;
  if (tid >= N * K) return;
  int e    = tid & 15;
  int lane = (tid >> 4) & 31;
  int blk  = tid >> 9;
  int kblks = K >> 5;
  int kblk = blk % kblks;
  int nblk = blk / kblks;
  int nn = nblk * 16 + (lane & 15);
  int kk = kblk * 32 + (lane >> 4) * 16 + e;
  out[tid] = (bf16)W[(size_t)nn * K + kk];
}

// ---------------------------------------------------------------------------
// Encoder: emb[t,b,:] = x[b,t,:] @ W_enc^T + b_enc   (bf16 out, f32 accum)
// One wave per 16(M) x 64(N) strip, K=512.  Fully unrolled so the scheduler
// overlaps global loads with WMMA issue (no rotating-register copies).
// ---------------------------------------------------------------------------
__global__ __launch_bounds__(256) void encoder_kernel(
    const bf16* __restrict__ xbf,     // [B,T,INDIM] bf16
    const bf16* __restrict__ WencS,   // swizzled [64 nblk][16 kblk][32][16]
    const float* __restrict__ b_enc,  // [EMB]
    bf16* __restrict__ emb)           // [T,B,EMB]
{
  const int wid  = blockIdx.x * 8 + (threadIdx.x >> 5);  // 0..32767
  const int lane = threadIdx.x & 31;
  const int ng   = wid & 15;   // 64-col N group
  const int mt   = wid >> 4;   // 0..2047 = t*16 + bt
  const int t    = mt >> 4;
  const int bt   = mt & 15;
  const int arow = lane & 15, half = lane >> 4;

  const bf16* xrow = xbf + ((size_t)(bt * 16 + arow) * TSTEPS + t) * INDIM;

  auto loadB = [&](int i, int kb) -> v16bf {
    int nblk = ng * 4 + i;
    size_t off = (((size_t)(nblk * (INDIM / 32) + kb)) * 32 + lane) * 16;
    V16 u;
    u.h[0] = *(const v8bf*)(WencS + off);
    u.h[1] = *(const v8bf*)(WencS + off + 8);
    return u.v;
  };

  v8f acc[4] = {{0,0,0,0,0,0,0,0},{0,0,0,0,0,0,0,0},
                {0,0,0,0,0,0,0,0},{0,0,0,0,0,0,0,0}};

#pragma unroll
  for (int kb = 0; kb < INDIM / 32; ++kb) {
    v16bf a  = load_a(xrow, kb * 32, half);
    v16bf b0 = loadB(0, kb);
    v16bf b1 = loadB(1, kb);
    v16bf b2 = loadB(2, kb);
    v16bf b3 = loadB(3, kb);
    acc[0] = wmma_bf16(a, b0, acc[0]);
    acc[1] = wmma_bf16(a, b1, acc[1]);
    acc[2] = wmma_bf16(a, b2, acc[2]);
    acc[3] = wmma_bf16(a, b3, acc[3]);
  }

  // C/D layout: lane holds column n=lane%16; VGPR r holds row r+8*(lane/16)
#pragma unroll
  for (int i = 0; i < 4; ++i) {
    int n = ng * 64 + i * 16 + (lane & 15);
    float bias = b_enc[n];
#pragma unroll
    for (int r = 0; r < 8; ++r) {
      int brow = bt * 16 + r + 8 * half;
      emb[((size_t)t * BSZ + brow) * EMB + n] = (bf16)(acc[i][r] + bias);
    }
  }
}

// ---------------------------------------------------------------------------
// Fused LSTM step for one (t, layer):
//   gates = x@Wih^T + h@Whh^T + b_ih + b_hh ; cell update ; h,c out.
// Grid: (64 n-tiles, 4 m-groups), block = 128 threads = 4 waves.
// Each wave owns a 16x16 h-tile with all 4 gate accumulators (i,f,g,o).
// Weight panel (8 strips: 4 gates x {Wih,Whh}) streamed into a 64KB LDS
// buffer via TDM in 4 K-phases; inner loop = ds_load + v_wmma only.
// ---------------------------------------------------------------------------
#define KPHASES 4
#define KB_PER_PHASE 8   // 8 kblocks * 32 = 256 K per phase

__global__ __launch_bounds__(128) void lstm_step_kernel(
    const bf16* __restrict__ xin,    // [B,HID] bf16
    const bf16* __restrict__ hprev,  // [B,HID] bf16
    const bf16* __restrict__ WihS,   // swizzled [256 nblk][32 kblk][32][16]
    const bf16* __restrict__ WhhS,
    const float* __restrict__ b_ih, const float* __restrict__ b_hh,
    float* __restrict__ c,           // [B,HID] f32, updated in place
    bf16* __restrict__ hout)         // [B,HID] bf16
{
  const int nt   = blockIdx.x;            // 0..63  (16 h-columns)
  const int mg   = blockIdx.y;            // 0..3   (64 batch rows)
  const int wave = threadIdx.x >> 5;      // 0..3
  const int lane = threadIdx.x & 31;
  const int mt   = mg * 4 + wave;         // 0..15
  const int m0   = mt * 16;
  const int arow = lane & 15, half = lane >> 4;

  // 8 frag-strips x 8 kblocks x 1KB = 64KB (single shared var -> LDS offset 0)
  __shared__ __align__(128) bf16 wlds[8][KB_PER_PHASE][512];

  const bf16* xrow = xin   + (size_t)(m0 + arow) * HID;
  const bf16* hrow = hprev + (size_t)(m0 + arow) * HID;

  v8f acc[4] = {{0,0,0,0,0,0,0,0},{0,0,0,0,0,0,0,0},
                {0,0,0,0,0,0,0,0},{0,0,0,0,0,0,0,0}};

  for (int p = 0; p < KPHASES; ++p) {
    if (wave == 0) {
      // 8 TDM 1-D strip loads: 8KB each (1024 x 8B elements), wave-uniform.
#pragma unroll
      for (int f = 0; f < 8; ++f) {
        int g = f & 3;
        const bf16* src = ((f >= 4) ? WhhS : WihS) +
            ((size_t)((g * 64 + nt) * (HID / 32) + p * KB_PER_PHASE)) * 512;
        tdm_load_strip(src, (unsigned)(f * KB_PER_PHASE * 1024),
                       KB_PER_PHASE * 128 /* 8B elems */);
      }
      __builtin_amdgcn_s_wait_tensorcnt(0);
    }
    // The TDM builtin carries no LDS pointer, so LLVM would otherwise treat
    // wlds as never-stored and fold the reads below to undef.  Escape the
    // LDS address into an opaque asm with a memory clobber: the compiler
    // must now assume wlds is (re)written each phase.
    {
      unsigned long long wp = (unsigned long long)(uintptr_t)&wlds[0][0][0];
      asm volatile("" : "+v"(wp) :: "memory");
    }
    __syncthreads();

#pragma unroll
    for (int kb = 0; kb < KB_PER_PHASE; ++kb) {
      v16bf aX = load_a(xrow, (p * KB_PER_PHASE + kb) * 32, half);
      v16bf aH = load_a(hrow, (p * KB_PER_PHASE + kb) * 32, half);
#pragma unroll
      for (int g = 0; g < 4; ++g) {
        v16bf bI = *(const v16bf*)&wlds[g][kb][lane * 16];
        acc[g] = wmma_bf16(aX, bI, acc[g]);
      }
#pragma unroll
      for (int g = 0; g < 4; ++g) {
        v16bf bH = *(const v16bf*)&wlds[4 + g][kb][lane * 16];
        acc[g] = wmma_bf16(aH, bH, acc[g]);
      }
    }
    __syncthreads();
  }

  // Epilogue: full LSTM cell update, gates never leave registers.
  const int n = nt * 16 + (lane & 15);
  float bias[4];
#pragma unroll
  for (int g = 0; g < 4; ++g) bias[g] = b_ih[g * HID + n] + b_hh[g * HID + n];

#pragma unroll
  for (int r = 0; r < 8; ++r) {
    int m = m0 + r + 8 * half;
    float ig = sigf(acc[0][r] + bias[0]);
    float fg = sigf(acc[1][r] + bias[1]);
    float gg = tanhf_fast(acc[2][r] + bias[2]);
    float og = sigf(acc[3][r] + bias[3]);
    size_t idx = (size_t)m * HID + n;
    float cc = fg * c[idx] + ig * gg;
    c[idx] = cc;
    hout[idx] = (bf16)(og * tanhf_fast(cc));
  }
}

// ---------------------------------------------------------------------------
// Classifier: y = h_last @ W_cls^T + b_cls   (tiny: 256x10x1024)
// ---------------------------------------------------------------------------
__global__ void cls_kernel(const bf16* __restrict__ h,
                           const float* __restrict__ Wc,
                           const float* __restrict__ bc,
                           float* __restrict__ y) {
  int tid = blockIdx.x * blockDim.x + threadIdx.x;
  if (tid >= BSZ * 10) return;
  int b = tid / 10, cl = tid % 10;
  const bf16* hr = h + (size_t)b * HID;
  const float* wr = Wc + (size_t)cl * HID;
  float s = 0.0f;
  for (int k = 0; k < HID; ++k) s += (float)hr[k] * wr[k];
  y[tid] = s + bc[cl];
}

// ---------------------------------------------------------------------------
extern "C" void kernel_launch(void* const* d_in, const int* in_sizes, int n_in,
                              void* d_out, int out_size, void* d_ws, size_t ws_size,
                              hipStream_t stream) {
  const float* x     = (const float*)d_in[0];
  const float* W_enc = (const float*)d_in[1];
  const float* b_enc = (const float*)d_in[2];
  const float* W_ih0 = (const float*)d_in[3];
  const float* W_hh0 = (const float*)d_in[4];
  const float* b_ih0 = (const float*)d_in[5];
  const float* b_hh0 = (const float*)d_in[6];
  const float* W_ih1 = (const float*)d_in[7];
  const float* W_hh1 = (const float*)d_in[8];
  const float* b_ih1 = (const float*)d_in[9];
  const float* b_hh1 = (const float*)d_in[10];
  const float* W_cls = (const float*)d_in[11];
  const float* b_cls = (const float*)d_in[12];

  char* ws = (char*)d_ws;
  size_t off = 0;
  auto alloc = [&](size_t bytes) {
    void* p = ws + off;
    off += (bytes + 255) & ~(size_t)255;
    return p;
  };

  bf16* xbf   = (bf16*)alloc((size_t)BSZ * TSTEPS * INDIM * 2);   // 32 MB
  bf16* emb   = (bf16*)alloc((size_t)TSTEPS * BSZ * EMB * 2);     // 64 MB
  bf16* WencS = (bf16*)alloc((size_t)EMB * INDIM * 2);            //  1 MB
  bf16* Wih0S = (bf16*)alloc((size_t)4 * HID * EMB * 2);          //  8 MB
  bf16* Whh0S = (bf16*)alloc((size_t)4 * HID * HID * 2);
  bf16* Wih1S = (bf16*)alloc((size_t)4 * HID * HID * 2);
  bf16* Whh1S = (bf16*)alloc((size_t)4 * HID * HID * 2);
  bf16* h0a   = (bf16*)alloc((size_t)BSZ * HID * 2);
  bf16* h0b   = (bf16*)alloc((size_t)BSZ * HID * 2);
  bf16* h1a   = (bf16*)alloc((size_t)BSZ * HID * 2);
  bf16* h1b   = (bf16*)alloc((size_t)BSZ * HID * 2);
  float* c0   = (float*)alloc((size_t)BSZ * HID * 4);
  float* c1   = (float*)alloc((size_t)BSZ * HID * 4);

  // deterministic init of recurrent state every call
  hipMemsetAsync(h0a, 0, (size_t)BSZ * HID * 2, stream);
  hipMemsetAsync(h1a, 0, (size_t)BSZ * HID * 2, stream);
  hipMemsetAsync(c0, 0, (size_t)BSZ * HID * 4, stream);
  hipMemsetAsync(c1, 0, (size_t)BSZ * HID * 4, stream);

  // precision prep
  {
    int n = BSZ * TSTEPS * INDIM;
    cvt_bf16_kernel<<<(n + 255) / 256, 256, 0, stream>>>(x, xbf, n);
  }
  swizzle_weight_kernel<<<(EMB * INDIM + 255) / 256, 256, 0, stream>>>(
      W_enc, WencS, EMB, INDIM);
  swizzle_weight_kernel<<<(4 * HID * EMB + 255) / 256, 256, 0, stream>>>(
      W_ih0, Wih0S, 4 * HID, EMB);
  swizzle_weight_kernel<<<(4 * HID * HID + 255) / 256, 256, 0, stream>>>(
      W_hh0, Whh0S, 4 * HID, HID);
  swizzle_weight_kernel<<<(4 * HID * HID + 255) / 256, 256, 0, stream>>>(
      W_ih1, Wih1S, 4 * HID, HID);
  swizzle_weight_kernel<<<(4 * HID * HID + 255) / 256, 256, 0, stream>>>(
      W_hh1, Whh1S, 4 * HID, HID);

  // encoder: 32768 waves = 4096 blocks of 8 waves
  encoder_kernel<<<4096, 256, 0, stream>>>(xbf, WencS, b_enc, emb);

  // sequential LSTM: 2 fused kernels per timestep
  bf16 *h0p = h0a, *h0n = h0b, *h1p = h1a, *h1n = h1b;
  for (int t = 0; t < TSTEPS; ++t) {
    lstm_step_kernel<<<dim3(64, 4), 128, 0, stream>>>(
        emb + (size_t)t * BSZ * EMB, h0p, Wih0S, Whh0S, b_ih0, b_hh0, c0, h0n);
    lstm_step_kernel<<<dim3(64, 4), 128, 0, stream>>>(
        h0n, h1p, Wih1S, Whh1S, b_ih1, b_hh1, c1, h1n);
    bf16* tmp;
    tmp = h0p; h0p = h0n; h0n = tmp;
    tmp = h1p; h1p = h1n; h1n = tmp;
  }

  // classifier on final h of layer 1 (h1p after the last swap)
  cls_kernel<<<(BSZ * 10 + 255) / 256, 256, 0, stream>>>(
      h1p, W_cls, b_cls, (float*)d_out);
}
